// TPAttention_39101382263291
// MI455X (gfx1250) — matrix operand order repaired and checked by reading, not compiled, not run on hardware
//
#include <hip/hip_runtime.h>
#include <hip/hip_bf16.h>

// ---------------------------------------------------------------------------
// TPA attention for MI455X (gfx1250, wave32, WMMA + TDM).
// ~130 GFLOP vs ~100MB HBM traffic -> matrix-core bound: projections, Q.K^T,
// P.V and the output projection all run on v_wmma_f32_16x16x32_bf16.
// Flash Q/K tiles are staged by the Tensor Data Mover (tensor_load_to_lds
// with LDS row padding), overlapped with manual V-transpose staging.
// ---------------------------------------------------------------------------

typedef __attribute__((ext_vector_type(16))) __bf16 bf16x16;
typedef __attribute__((ext_vector_type(8)))  __bf16 bf16x8;
typedef __attribute__((ext_vector_type(8)))  float  f32x8;
typedef __attribute__((ext_vector_type(4)))  unsigned int u32x4;
typedef __attribute__((ext_vector_type(8)))  int i32x8;
typedef __attribute__((ext_vector_type(4)))  int i32x4;

union FragA { bf16x16 v; bf16x8 h[2]; };

__device__ __forceinline__ f32x8 wmma_bf16(const bf16x16& a, const bf16x16& b, const f32x8& c) {
    return __builtin_amdgcn_wmma_f32_16x16x32_bf16(false, a, false, b, (short)0, c, false, false);
}

// Problem constants
#define BATCH   2
#define SEQ     2048
#define DMODEL  2048
#define NHEAD   16
#define DHEAD   128
#define CHN     144      // NHEAD + DHEAD
#define RQ      6
#define RK      2
#define RV      2
#define NROWS   (BATCH*SEQ)   // 4096

// ---------------------------------------------------------------------------
// TDM 2D tile load: rows x 128 bf16, row stride 2048 bf16 in memory,
// LDS stride 136 bf16 (64 DWORDs data + 4 DWORDs pad per row).
// Descriptor per CDNA5 ISA.
// This toolchain exposes the 6-arg clang-23 builtin:
//   (u32x4 g0, i32x8 g1, i32x4 g2, i32x4 g3, i32x8 g4, i32 cpol)
// ---------------------------------------------------------------------------
__device__ __forceinline__ void tdm_load_rows128(unsigned int lds_byte_off,
                                                 const __bf16* gptr,
                                                 unsigned int rows)
{
    unsigned long long ga = (unsigned long long)(uintptr_t)gptr;
    u32x4 g0;
    g0[0] = 1u;                                           // count=1, user desc
    g0[1] = lds_byte_off;                                 // lds_addr
    g0[2] = (unsigned int)ga;                             // global_addr[31:0]
    g0[3] = ((unsigned int)(ga >> 32) & 0x01FFFFFFu) | (2u << 30); // addr[56:32] | type=2
    i32x8 g1;
    g1[0] = (int)((1u << 16)        // data_size = 2 bytes
                | (1u << 20)        // pad_enable
                | (5u << 22)        // pad_interval: 64 DWORDs (2^(5+1))
                | (3u << 25));      // pad_amount: 4 DWORDs
    g1[1] = (int)(128u << 16);                  // tensor_dim0 = 128 (lo16)
    g1[2] = (int)((unsigned int)SEQ << 16);     // dim0 hi=0 | tensor_dim1 lo16 = SEQ
    g1[3] = (int)(128u << 16);                  // dim1 hi=0 | tile_dim0 = 128
    g1[4] = (int)rows;                          // tile_dim1 = rows | tile_dim2 = 0
    g1[5] = (int)(NHEAD * DHEAD);               // tensor_dim0_stride = 2048 (lo32)
    g1[6] = 0;                                  // stride hi | dim1_stride (unused, 2D)
    g1[7] = 0;
    i32x4 z4 = {0, 0, 0, 0};
    i32x8 z8 = {0, 0, 0, 0, 0, 0, 0, 0};
    __builtin_amdgcn_tensor_load_to_lds(g0, g1, z4, z4, z8, 0);
}

// ---------------------------------------------------------------------------
// Kernel 1/5: C[M,N] = A[M,K] * B[N,K]^T   (A,B fp32 row-major, C fp32)
// Block tile 128(M) x 64(N), K-step 32, double-buffered LDS.
// 8 waves, each wave owns 32x32 (4 WMMAs per K-step).
// Requires: M % 128 == 0, K % 32 == 0 (true for all uses; N is guarded).
// ---------------------------------------------------------------------------
#define GBM 128
#define GBN 64
#define GBK 32
#define GLS 40   // LDS row stride in elements (80B, keeps 16B alignment)

__global__ __launch_bounds__(256)
void tpa_gemm_nt(const float* __restrict__ A, const float* __restrict__ B,
                 float* __restrict__ C, int M, int N, int K)
{
    __shared__ __align__(16) __bf16 As[2][GBM * GLS];
    __shared__ __align__(16) __bf16 Bs[2][GBN * GLS];

    const int tid  = threadIdx.x;
    const int lane = tid & 31;
    const int w    = tid >> 5;        // 0..7
    const int wm   = w >> 1;          // 0..3 (M)
    const int wn   = w & 1;           // 0..1 (N)
    const int bm0  = blockIdx.y * GBM;
    const int bn0  = blockIdx.x * GBN;

    f32x8 acc[2][2] = {};
    float4 ra[4], rb[2];

    // batch-issue all global loads of one K-tile into registers (MLP!)
    auto load_tiles = [&](int k0) {
        #pragma unroll
        for (int i = 0; i < 4; ++i) {
            int e = tid + i * 256;              // 0..1023
            int row = e >> 3, ch = e & 7;
            ra[i] = *(const float4*)&A[(size_t)(bm0 + row) * K + k0 + ch * 4];
        }
        #pragma unroll
        for (int i = 0; i < 2; ++i) {
            int e = tid + i * 256;              // 0..511
            int row = e >> 3, ch = e & 7;
            int n = bn0 + row;
            float4 f = {0.f, 0.f, 0.f, 0.f};
            if (n < N) f = *(const float4*)&B[(size_t)n * K + k0 + ch * 4];
            rb[i] = f;
        }
    };
    // convert fp32->bf16 and store staged registers to LDS buffer `buf`
    auto store_tiles = [&](int buf) {
        #pragma unroll
        for (int i = 0; i < 4; ++i) {
            int e = tid + i * 256;
            int row = e >> 3, ch = e & 7;
            __bf16* dst = &As[buf][row * GLS + ch * 4];
            dst[0] = (__bf16)ra[i].x; dst[1] = (__bf16)ra[i].y;
            dst[2] = (__bf16)ra[i].z; dst[3] = (__bf16)ra[i].w;
        }
        #pragma unroll
        for (int i = 0; i < 2; ++i) {
            int e = tid + i * 256;
            int row = e >> 3, ch = e & 7;
            __bf16* dst = &Bs[buf][row * GLS + ch * 4];
            dst[0] = (__bf16)rb[i].x; dst[1] = (__bf16)rb[i].y;
            dst[2] = (__bf16)rb[i].z; dst[3] = (__bf16)rb[i].w;
        }
    };

    load_tiles(0);
    store_tiles(0);
    const int nk = K / GBK;

    for (int kt = 0; kt < nk; ++kt) {
        __syncthreads();
        if (kt + 1 < nk) load_tiles((kt + 1) * GBK);   // overlap with WMMA below

        const __bf16* as = As[kt & 1];
        const __bf16* bs = Bs[kt & 1];
        const int aK0 = (lane >= 16) ? 8 : 0;       // CDNA5 16-bit A layout
        const int bK0 = (lane >= 16) ? 8 : 0;       // CDNA5 16-bit B layout
        const int ar  = wm * 32 + (lane & 15);
        const int bc  = wn * 32 + (lane & 15);
        FragA a0, a1, b0, b1;
        a0.h[0] = *(const bf16x8*)&as[ar * GLS + aK0];
        a0.h[1] = *(const bf16x8*)&as[ar * GLS + aK0 + 16];
        a1.h[0] = *(const bf16x8*)&as[(ar + 16) * GLS + aK0];
        a1.h[1] = *(const bf16x8*)&as[(ar + 16) * GLS + aK0 + 16];
        b0.h[0] = *(const bf16x8*)&bs[bc * GLS + bK0];
        b0.h[1] = *(const bf16x8*)&bs[bc * GLS + bK0 + 16];
        b1.h[0] = *(const bf16x8*)&bs[(bc + 16) * GLS + bK0];
        b1.h[1] = *(const bf16x8*)&bs[(bc + 16) * GLS + bK0 + 16];

        acc[0][0] = wmma_bf16(a0.v, b0.v, acc[0][0]);
        acc[0][1] = wmma_bf16(a0.v, b1.v, acc[0][1]);
        acc[1][0] = wmma_bf16(a1.v, b0.v, acc[1][0]);
        acc[1][1] = wmma_bf16(a1.v, b1.v, acc[1][1]);

        if (kt + 1 < nk) store_tiles((kt + 1) & 1);    // other buffer: no race
    }

    const int roff = (lane >= 16) ? 8 : 0;
    const int col  = lane & 15;
    #pragma unroll
    for (int i = 0; i < 2; ++i)
        #pragma unroll
        for (int j = 0; j < 2; ++j)
            #pragma unroll
            for (int r = 0; r < 8; ++r) {
                int m = bm0 + wm * 32 + i * 16 + r + roff;
                int n = bn0 + wn * 32 + j * 16 + col;
                if (n < N) C[(size_t)m * N + n] = acc[i][j][r];
            }
}

// ---------------------------------------------------------------------------
// Kernel 3: per (b,t) rotary on b_kv + rank contraction -> q,k,v (bf16)
// ---------------------------------------------------------------------------
__global__ __launch_bounds__(128)
void tpa_qkv_form(const float* __restrict__ abq, const float* __restrict__ abkv,
                  __bf16* __restrict__ q, __bf16* __restrict__ k, __bf16* __restrict__ v)
{
    __shared__ float aq[RQ * CHN];              // 864 fp32
    __shared__ float akv[(RK + RV) * CHN];      // 576 fp32

    const int bt  = blockIdx.x;
    const int tid = threadIdx.x;

    for (int i = tid; i < RQ * CHN; i += 128)        aq[i]  = abq[(size_t)bt * (RQ * CHN) + i];
    for (int i = tid; i < (RK + RV) * CHN; i += 128) akv[i] = abkv[(size_t)bt * ((RK + RV) * CHN) + i];
    __syncthreads();

    const int t = bt & (SEQ - 1);
    for (int e = tid; e < (RK + RV) * (DHEAD / 2); e += 128) {   // 4*64 rotary pairs
        int r = e >> 6, j = e & 63;
        float inv = __expf(-9.210340371976184f * (float)j * (1.0f / 64.0f)); // 10000^(-j/64)
        float f   = (float)t * inv;
        float sn, cs; __sincosf(f, &sn, &cs);
        float x1 = akv[r * CHN + 16 + j];
        float x2 = akv[r * CHN + 16 + 64 + j];
        akv[r * CHN + 16 + j]      =  x1 * cs + x2 * sn;
        akv[r * CHN + 16 + 64 + j] = -x1 * sn + x2 * cs;
    }
    __syncthreads();

    for (int e = tid; e < NHEAD * DHEAD; e += 128) {
        int hh = e >> 7, d = e & 127;
        float sq = 0.f;
        #pragma unroll
        for (int r = 0; r < RQ; ++r) sq += aq[r * CHN + hh] * aq[r * CHN + 16 + d];
        float sk = 0.f, sv = 0.f;
        #pragma unroll
        for (int r = 0; r < RK; ++r) sk += akv[r * CHN + hh] * akv[r * CHN + 16 + d];
        #pragma unroll
        for (int r = RK; r < RK + RV; ++r) sv += akv[r * CHN + hh] * akv[r * CHN + 16 + d];
        size_t o = (size_t)bt * (NHEAD * DHEAD) + e;
        q[o] = (__bf16)(sq * (1.0f / RQ));
        k[o] = (__bf16)(sk * (1.0f / RK));
        v[o] = (__bf16)(sv * (1.0f / RV));
    }
}

// ---------------------------------------------------------------------------
// Kernel 4: causal flash attention, WMMA for S = Q.K^T and O += P.V.
// Q and K tiles staged by TDM (tensor_load_to_lds + s_wait_tensorcnt),
// V staged transposed by VALU/DS, overlapped with the TDM DMA.
// Block: 128 threads (4 waves).  Q tile 64 rows, K tiles of 32.
// Dynamic LDS layout (byte offsets are TDM lds_addr values):
//   Qs @ 0       : 64*136 bf16 = 17408B
//   Ks @ 17408   : 32*136 bf16 =  8704B
//   Vt @ 26112   : 128*40 bf16 = 10240B   (V transposed [d][k])
//   Ps @ 36352   : 4*16*40 bf16 = 5120B   (per-wave P tile)
// ---------------------------------------------------------------------------
#define FLASH_LDS_BYTES 41472
#define KS_OFF 17408

__global__ __launch_bounds__(128)
void tpa_flash(const __bf16* __restrict__ qb, const __bf16* __restrict__ kb,
               const __bf16* __restrict__ vb, float* __restrict__ y)
{
    extern __shared__ char smem_raw[];
    __bf16* Qs = (__bf16*)smem_raw;                    // stride 136
    __bf16* Ks = (__bf16*)(smem_raw + KS_OFF);         // stride 136
    __bf16* Vt = (__bf16*)(smem_raw + 26112);          // stride 40
    __bf16* Ps = (__bf16*)(smem_raw + 36352);          // per-wave 640 elems

    const int bh = blockIdx.x;                 // b*NHEAD + h
    const int b  = bh >> 4;
    const int h  = bh & 15;
    const int qt = blockIdx.y;
    const int q0 = qt * 64;

    const int tid  = threadIdx.x;
    const int lane = tid & 31;
    const int w    = tid >> 5;                 // wave -> 16 rows
    const int col  = lane & 15;
    const int roff = (lane >= 16) ? 8 : 0;

    const size_t head_base = ((size_t)b * SEQ) * (NHEAD * DHEAD) + (size_t)h * DHEAD;

    // ---- TDM: stage Q tile (64x128 bf16, padded rows) ----
    if (w == 0) {
        tdm_load_rows128(0u, qb + head_base + (size_t)q0 * (NHEAD * DHEAD), 64u);
        __builtin_amdgcn_s_wait_tensorcnt(0);
    }
    __syncthreads();

    // ---- Q A-fragments held in registers for the whole loop ----
    FragA qa[4];
    {
        const int m = w * 16 + col;
        #pragma unroll
        for (int ks = 0; ks < 4; ++ks) {
            qa[ks].h[0] = *(const bf16x8*)&Qs[m * 136 + ks * 32 + roff];
            qa[ks].h[1] = *(const bf16x8*)&Qs[m * 136 + ks * 32 + roff + 16];
        }
    }

    f32x8 o[8] = {};                            // 16x128 output per wave
    float row_m[8], row_l[8];
    #pragma unroll
    for (int r = 0; r < 8; ++r) { row_m[r] = -3.0e38f; row_l[r] = 0.f; }

    const float sc  = 0.08838834764831845f;     // 1/sqrt(128)
    const int   nkt = 2 * qt + 2;               // causal tile count

    for (int kt = 0; kt < nkt; ++kt) {
        __syncthreads();
        // ---- TDM: stage K tile (32x128), async while we stage V below ----
        if (w == 0)
            tdm_load_rows128(KS_OFF, kb + head_base + (size_t)(kt * 32) * (NHEAD * DHEAD), 32u);

        // ---- stage V transposed (Vt[d][k]); batched uint2 loads ----
        uint2 vr[8];
        #pragma unroll
        for (int i = 0; i < 8; ++i) {
            int e  = tid + i * 128;             // 0..1023 chunks of 4 bf16
            int kk = e >> 5;                    // 0..31
            int c4 = e & 31;                    // chunk along d
            vr[i] = *(const uint2*)&vb[head_base + (size_t)(kt * 32 + kk) * (NHEAD * DHEAD) + c4 * 4];
        }
        #pragma unroll
        for (int i = 0; i < 8; ++i) {
            int e  = tid + i * 128;
            int kk = e >> 5;
            int c4 = e & 31;
            const __bf16* p = (const __bf16*)&vr[i];
            #pragma unroll
            for (int j = 0; j < 4; ++j) Vt[(c4 * 4 + j) * 40 + kk] = p[j];
        }
        if (w == 0) __builtin_amdgcn_s_wait_tensorcnt(0);
        __syncthreads();

        // ---- S = Q K^T (16x32 per wave) ----
        f32x8 s0 = {}, s1 = {};
        const int bK0 = (lane >= 16) ? 8 : 0;
        #pragma unroll
        for (int ks = 0; ks < 4; ++ks) {
            FragA k0f, k1f;
            k0f.h[0] = *(const bf16x8*)&Ks[col * 136 + ks * 32 + bK0];
            k0f.h[1] = *(const bf16x8*)&Ks[col * 136 + ks * 32 + bK0 + 16];
            k1f.h[0] = *(const bf16x8*)&Ks[(16 + col) * 136 + ks * 32 + bK0];
            k1f.h[1] = *(const bf16x8*)&Ks[(16 + col) * 136 + ks * 32 + bK0 + 16];
            s0 = wmma_bf16(qa[ks].v, k0f.v, s0);
            s1 = wmma_bf16(qa[ks].v, k1f.v, s1);
        }

        // ---- online softmax (rows live in 16-lane halves -> xor reduce) ----
        #pragma unroll
        for (int r = 0; r < 8; ++r) {
            float e0 = s0[r] * sc, e1 = s1[r] * sc;
            int qrow = q0 + w * 16 + r + roff;
            if (kt * 32 + col      > qrow) e0 = -3.0e38f;
            if (kt * 32 + 16 + col > qrow) e1 = -3.0e38f;
            float mx = fmaxf(e0, e1);
            #pragma unroll
            for (int msk = 1; msk < 16; msk <<= 1) mx = fmaxf(mx, __shfl_xor(mx, msk, 32));
            float nm   = fmaxf(row_m[r], mx);
            float corr = __expf(row_m[r] - nm);
            float p0 = __expf(e0 - nm);
            float p1 = __expf(e1 - nm);
            float ps = p0 + p1;
            #pragma unroll
            for (int msk = 1; msk < 16; msk <<= 1) ps += __shfl_xor(ps, msk, 32);
            row_l[r] = row_l[r] * corr + ps;
            row_m[r] = nm;
            #pragma unroll
            for (int dt = 0; dt < 8; ++dt) o[dt][r] *= corr;
            Ps[w * 640 + (r + roff) * 40 + col]      = (__bf16)p0;
            Ps[w * 640 + (r + roff) * 40 + col + 16] = (__bf16)p1;
        }

        // ---- O += P V  (P: 16x32 A fragment, V: 32x128) ----
        FragA pa;
        pa.h[0] = *(const bf16x8*)&Ps[w * 640 + col * 40 + roff];
        pa.h[1] = *(const bf16x8*)&Ps[w * 640 + col * 40 + roff + 16];
        const int vK0 = (lane >= 16) ? 8 : 0;
        #pragma unroll
        for (int dt = 0; dt < 8; ++dt) {
            FragA vf;
            vf.h[0] = *(const bf16x8*)&Vt[(dt * 16 + col) * 40 + vK0];
            vf.h[1] = *(const bf16x8*)&Vt[(dt * 16 + col) * 40 + vK0 + 16];
            o[dt] = wmma_bf16(pa.v, vf.v, o[dt]);
        }
    }

    // ---- normalize + store y (fp32, [b,t,h*128+d]) ----
    #pragma unroll
    for (int r = 0; r < 8; ++r) {
        float inv_l = 1.0f / row_l[r];
        int trow = q0 + w * 16 + r + roff;
        size_t base = ((size_t)b * SEQ + trow) * (NHEAD * DHEAD) + (size_t)h * DHEAD;
        #pragma unroll
        for (int dt = 0; dt < 8; ++dt)
            y[base + dt * 16 + col] = o[dt][r] * inv_l;
    }
}

// ---------------------------------------------------------------------------
// Launch: x, W_q, W_kv, W_o  ->  out (B,T,D) fp32
// Workspace: abq(14.2MB) | abkv(9.4MB) | q,k,v bf16(3x16.8MB) | y fp32(33.6MB)
// ---------------------------------------------------------------------------
extern "C" void kernel_launch(void* const* d_in, const int* in_sizes, int n_in,
                              void* d_out, int out_size, void* d_ws, size_t ws_size,
                              hipStream_t stream)
{
    const float* x   = (const float*)d_in[0];
    const float* Wq  = (const float*)d_in[1];
    const float* Wkv = (const float*)d_in[2];
    const float* Wo  = (const float*)d_in[3];
    float* out = (float*)d_out;

    char* ws = (char*)d_ws;
    const size_t SZ_ABQ  = (size_t)NROWS * (RQ * CHN) * sizeof(float);
    const size_t SZ_ABKV = (size_t)NROWS * ((RK + RV) * CHN) * sizeof(float);
    const size_t SZ_HD   = (size_t)NROWS * (NHEAD * DHEAD);
    float*  abq  = (float*)(ws);
    float*  abkv = (float*)(ws + SZ_ABQ);
    __bf16* qbuf = (__bf16*)(ws + SZ_ABQ + SZ_ABKV);
    __bf16* kbuf = qbuf + SZ_HD;
    __bf16* vbuf = kbuf + SZ_HD;
    float*  ybuf = (float*)((char*)(vbuf + SZ_HD));

    // 1) abq = x @ W_q^T        (4096 x 864)
    tpa_gemm_nt<<<dim3((RQ * CHN + GBN - 1) / GBN, NROWS / GBM), 256, 0, stream>>>(
        x, Wq, abq, NROWS, RQ * CHN, DMODEL);
    // 2) abkv = x @ W_kv^T      (4096 x 576)
    tpa_gemm_nt<<<dim3(((RK + RV) * CHN + GBN - 1) / GBN, NROWS / GBM), 256, 0, stream>>>(
        x, Wkv, abkv, NROWS, (RK + RV) * CHN, DMODEL);
    // 3) rotary + rank contraction -> q,k,v (bf16)
    tpa_qkv_form<<<NROWS, 128, 0, stream>>>(abq, abkv, qbuf, kbuf, vbuf);
    // 4) causal flash attention -> y (fp32)
    tpa_flash<<<dim3(BATCH * NHEAD, SEQ / 64), 128, FLASH_LDS_BYTES, stream>>>(
        qbuf, kbuf, vbuf, ybuf);
    // 5) out = y @ W_o^T        (4096 x 2048)
    tpa_gemm_nt<<<dim3(DMODEL / GBN, NROWS / GBM), 256, 0, stream>>>(
        ybuf, Wo, out, NROWS, DMODEL, DMODEL);
}
